// Contrast_IB_52458730553641
// MI455X (gfx1250) — matrix operand order repaired and checked by reading, not compiled, not run on hardware
//
#include <hip/hip_runtime.h>
#include <hip/hip_fp16.h>

typedef __attribute__((ext_vector_type(16))) _Float16 v16h;
typedef __attribute__((ext_vector_type(8)))  float    v8f;

#define GATE_BIAS_F 1e-4f
#define INV_TMP     5.0f   // 1/0.2

// ---------------- helpers ----------------

__device__ __forceinline__ float gather2(const float* __restrict__ a,
                                         const float* __restrict__ b,
                                         int U, int node, int c) {
  return (node < U) ? a[(size_t)node * 64 + c] : b[(size_t)(node - U) * 64 + c];
}

__device__ __forceinline__ float hash_u01(unsigned x, unsigned tag) {
  unsigned h = x * 2654435761u + tag * 0x9E3779B9u;
  h ^= h >> 16; h *= 0x85EBCA6Bu;
  h ^= h >> 13; h *= 0xC2B2AE35u;
  h ^= h >> 16;
  return (float)(h >> 8) * (1.0f / 16777216.0f) + (0.5f / 16777216.0f);
}

__device__ __forceinline__ float gumbel_gate(float logit, float u) {
  float eps = (GATE_BIAS_F - (1.0f - GATE_BIAS_F)) * u + (1.0f - GATE_BIAS_F);
  float z = (logf(eps) - logf(1.0f - eps) + logit) * INV_TMP;
  return 1.0f / (1.0f + expf(-z));
}

// ---------------- WMMA GEMM: Y = act(cur @ W + b) ----------------
// 64x64 output tile per workgroup (128 thr = 4 waves). Wave w owns rows
// [16w,16w+16) and iterates the 4 column tiles: 8 x v_wmma_f32_16x16x32_f16.

#define AS_STRIDE 66
#define WS_STRIDE 66

template <bool BIAS, bool RELU>
__launch_bounds__(128)
__global__ void gemm_cur64(const float* __restrict__ ue, const float* __restrict__ ie,
                           int U, int Nn,
                           const float* __restrict__ W,      // [64,64] row-major (k,n)
                           const float* __restrict__ bias,   // [64] (used iff BIAS)
                           float* __restrict__ Y) {
  __shared__ _Float16 As[64 * AS_STRIDE];
  __shared__ _Float16 Ws[64 * WS_STRIDE];
  const int tid  = threadIdx.x;
  const int base = blockIdx.x * 64;

  for (int i = tid; i < 64 * 64; i += 128) {
    int r = i >> 6, k = i & 63;
    float v = (base + r < Nn) ? gather2(ue, ie, U, base + r, k) : 0.0f;
    As[r * AS_STRIDE + k] = (_Float16)v;
    Ws[r * WS_STRIDE + k] = (_Float16)W[i];
  }
  __syncthreads();

  const int wave = tid >> 5, lane = tid & 31;
  const int m = (wave << 4) + (lane & 15);     // this wave's row within the 64-row tile
  const int half = lane >> 4;

  v16h a0, a1;
#pragma unroll
  for (int v = 0; v < 8; ++v) {
    int k0 = ((v >> 2) << 4) + (half << 3) + ((v & 3) << 1);   // ISA 16-bit A 16x32 layout
    a0[2 * v]     = As[m * AS_STRIDE + k0];
    a0[2 * v + 1] = As[m * AS_STRIDE + k0 + 1];
    a1[2 * v]     = As[m * AS_STRIDE + 32 + k0];
    a1[2 * v + 1] = As[m * AS_STRIDE + 32 + k0 + 1];
  }

#pragma unroll
  for (int ct = 0; ct < 4; ++ct) {
    const int col0 = ct * 16;
    v16h b0, b1;
#pragma unroll
    for (int j = 0; j < 16; ++j) {
      b0[j] = Ws[(((j >> 3) << 4) + (half << 3) + (j & 7)) * WS_STRIDE + col0 + (lane & 15)];          // lane = N, elem j = K (table order, 0..31)
      b1[j] = Ws[(32 + ((j >> 3) << 4) + (half << 3) + (j & 7)) * WS_STRIDE + col0 + (lane & 15)];   // lane = N, elem j = K (table order, 32..63)
    }
    v8f c = {};
    c = __builtin_amdgcn_wmma_f32_16x16x32_f16(false, a0, false, b0, (short)0, c, false, false);
    c = __builtin_amdgcn_wmma_f32_16x16x32_f16(false, a1, false, b1, (short)0, c, false, false);

    const int n  = col0 + (lane & 15);
    const float bn = BIAS ? bias[n] : 0.0f;
    const int rb = base + (wave << 4) + ((lane >> 4) << 3);   // row base for this lane's Cs
#pragma unroll
    for (int v = 0; v < 8; ++v) {
      float val = c[v] + bn;
      if (RELU) val = fmaxf(val, 0.0f);
      if (rb + v < Nn) Y[(size_t)(rb + v) * 64 + n] = val;
    }
  }
}

// ---------------- edge gate: ne[e] = adj[e] * gate(relu(P[row]+Q[col]+b1) . w2 + b2) ----------

__launch_bounds__(256)
__global__ void edge_gate_k(const float* __restrict__ P, const float* __restrict__ Q,
                            const float* __restrict__ b1, const float* __restrict__ w2,
                            const float* __restrict__ b2,
                            const int* __restrict__ row, const int* __restrict__ col,
                            const float* __restrict__ adj,
                            int E, unsigned tag,
                            float* __restrict__ ne) {
  int e = blockIdx.x * 8 + (threadIdx.x >> 5);
  int lane = threadIdx.x & 31;
  if (e >= E) return;
  int r = row[e], cc = col[e];
  float acc = 0.0f;
#pragma unroll
  for (int t = 0; t < 2; ++t) {
    int ch = lane + 32 * t;
    float h = P[(size_t)r * 64 + ch] + Q[(size_t)cc * 64 + ch] + b1[ch];
    acc += fmaxf(h, 0.0f) * w2[ch];
  }
#pragma unroll
  for (int off = 16; off > 0; off >>= 1) acc += __shfl_down(acc, off, 32);
  if (lane == 0) {
    float g = gumbel_gate(acc + b2[0], hash_u01((unsigned)e, tag));
    ne[e] = adj[e] * g;
  }
}

// ---------------- node gate: m[i] = gate(HN[i] . w2 + b2) ----------------

__launch_bounds__(256)
__global__ void node_gate_k(const float* __restrict__ HN,
                            const float* __restrict__ w2, const float* __restrict__ b2,
                            int Nn, unsigned tag, float* __restrict__ nm) {
  int i = blockIdx.x * 8 + (threadIdx.x >> 5);
  int lane = threadIdx.x & 31;
  if (i >= Nn) return;
  float acc = 0.0f;
#pragma unroll
  for (int t = 0; t < 2; ++t) {
    int ch = lane + 32 * t;
    acc += HN[(size_t)i * 64 + ch] * w2[ch];
  }
#pragma unroll
  for (int off = 16; off > 0; off >>= 1) acc += __shfl_down(acc, off, 32);
  if (lane == 0)
    nm[i] = gumbel_gate(acc + b2[0], hash_u01((unsigned)i, tag));
}

// ---------------- scatter: XN[row] += w[e] * X[col]  (HW f32 atomics) ----------------

__launch_bounds__(256)
__global__ void scatter_k(const int* __restrict__ row, const int* __restrict__ col,
                          const float* __restrict__ w,
                          const float* __restrict__ X, float* __restrict__ XN, int E) {
  int e = blockIdx.x * 8 + (threadIdx.x >> 5);
  int lane = threadIdx.x & 31;
  if (e >= E) return;
  int r = row[e], cc = col[e];
  // warm the atomic-destination line (global_prefetch_b8)
  __builtin_prefetch(&XN[(size_t)r * 64 + lane], 1, 3);
  float we = w[e];
#pragma unroll
  for (int t = 0; t < 2; ++t) {
    int ch = lane + 32 * t;
    unsafeAtomicAdd(&XN[(size_t)r * 64 + ch], we * X[(size_t)cc * 64 + ch]);
  }
}

// ---------------- RW mean-pool + gated mix: OUT = m*X + (1-m)*pool(X) ----------------

__launch_bounds__(256)
__global__ void pool_mix_k(const float* __restrict__ X,
                           const int* __restrict__ rw_cols, const float* __restrict__ rw_vals,
                           const float* __restrict__ nm, int Nn, int K,
                           float* __restrict__ OUT) {
  int i = blockIdx.x * 8 + (threadIdx.x >> 5);
  int lane = threadIdx.x & 31;
  if (i >= Nn) return;
  float m = nm[i];
  int ch0 = lane, ch1 = lane + 32;
  float s0 = 0.0f, s1 = 0.0f;
  for (int k = 0; k < K; ++k) {
    int   c = rw_cols[(size_t)i * K + k];
    float v = rw_vals[(size_t)i * K + k];
    s0 += v * X[(size_t)c * 64 + ch0];
    s1 += v * X[(size_t)c * 64 + ch1];
  }
  OUT[(size_t)i * 64 + ch0] = m * X[(size_t)i * 64 + ch0] + (1.0f - m) * s0;
  OUT[(size_t)i * 64 + ch1] = m * X[(size_t)i * 64 + ch1] + (1.0f - m) * s1;
}

// ---------------- elementwise kernels ----------------

__global__ void init_out_k(const float* __restrict__ ou, const float* __restrict__ oi,
                           int U, int Nn, float* __restrict__ out, float* __restrict__ X) {
  size_t idx = (size_t)blockIdx.x * blockDim.x + threadIdx.x;
  size_t tot = (size_t)Nn * 64;
  if (idx < tot) {
    int i = (int)(idx >> 6), c = (int)(idx & 63);
    float o = gather2(ou, oi, U, i, c);
    out[(size_t)i * 128 + c]      = o;   // edge view acc
    out[(size_t)i * 128 + 64 + c] = o;   // node view acc
    X[idx] = o;
  }
  if (idx == 0) { out[tot * 2] = 0.0f; out[tot * 2 + 1] = 0.0f; }
}

__global__ void copy_orig_k(const float* __restrict__ ou, const float* __restrict__ oi,
                            int U, int Nn, float* __restrict__ X) {
  size_t idx = (size_t)blockIdx.x * blockDim.x + threadIdx.x;
  if (idx >= (size_t)Nn * 64) return;
  int i = (int)(idx >> 6), c = (int)(idx & 63);
  X[idx] = gather2(ou, oi, U, i, c);
}

__global__ void zero_k(float* __restrict__ p, size_t n) {
  size_t idx = (size_t)blockIdx.x * blockDim.x + threadIdx.x;
  if (idx < n) p[idx] = 0.0f;
}

__global__ void accum_k(float* __restrict__ out, const float* __restrict__ XN,
                        int Nn, int off) {
  size_t idx = (size_t)blockIdx.x * blockDim.x + threadIdx.x;
  if (idx >= (size_t)Nn * 64) return;
  int i = (int)(idx >> 6), c = (int)(idx & 63);
  out[(size_t)i * 128 + off + c] += XN[idx];
}

__global__ void scale_k(float* __restrict__ out, size_t n, float s) {
  size_t idx = (size_t)blockIdx.x * blockDim.x + threadIdx.x;
  if (idx < n) out[idx] *= s;
}

__launch_bounds__(256)
__global__ void reduce_sum_k(const float* __restrict__ v, size_t n, float scale,
                             float* __restrict__ target) {
  __shared__ float sm[256];
  float s = 0.0f;
  for (size_t i = (size_t)blockIdx.x * 256 + threadIdx.x; i < n; i += (size_t)gridDim.x * 256)
    s += v[i];
  sm[threadIdx.x] = s;
  __syncthreads();
  for (int w = 128; w > 0; w >>= 1) {
    if ((int)threadIdx.x < w) sm[threadIdx.x] += sm[threadIdx.x + w];
    __syncthreads();
  }
  if (threadIdx.x == 0) unsafeAtomicAdd(target, sm[0] * scale);
}

// ---------------- host orchestration ----------------

extern "C" void kernel_launch(void* const* d_in, const int* in_sizes, int n_in,
                              void* d_out, int out_size, void* d_ws, size_t ws_size,
                              hipStream_t stream) {
  const float* cur_u  = (const float*)d_in[0];
  const float* cur_i  = (const float*)d_in[1];
  const float* orig_u = (const float*)d_in[2];
  const float* orig_i = (const float*)d_in[3];
  const int*   row    = (const int*)d_in[4];
  const int*   col    = (const int*)d_in[5];
  const float* adj    = (const float*)d_in[6];
  const int*   rw_cols = (const int*)d_in[8];
  const float* rw_vals = (const float*)d_in[9];
  const float* edge_w1 = (const float*)d_in[10];
  const float* edge_b1 = (const float*)d_in[11];
  const float* edge_w2 = (const float*)d_in[12];
  const float* edge_b2 = (const float*)d_in[13];
  const float* node_w1 = (const float*)d_in[14];
  const float* node_b1 = (const float*)d_in[15];
  const float* node_w2 = (const float*)d_in[16];
  const float* node_b2 = (const float*)d_in[17];

  const int U  = in_sizes[0] / 64;
  const int I  = in_sizes[1] / 64;
  const int Nn = U + I;
  const int E  = in_sizes[4];
  const int L  = in_sizes[13];          // edge_b2 has L elements
  const int K  = in_sizes[8] / Nn;      // walk fan-out (8)
  float* out = (float*)d_out;

  // workspace layout (floats)
  const size_t ND = (size_t)Nn * 64;
  float* P  = (float*)d_ws;
  float* Q  = P + ND;
  float* X  = Q + ND;
  float* XN = X + ND;
  float* MP = XN + ND;
  float* NE = MP + ND;                  // [L][E]
  float* NM = NE + (size_t)L * E;       // [L][N]

  const int TB = 256;
  const int gN64  = (int)((ND + TB - 1) / TB);
  const int gN128 = (int)((ND * 2 + TB - 1) / TB);
  const int gEdge = (E + 7) / 8;
  const int gNode = (Nn + 7) / 8;
  const int gTile = (Nn + 63) / 64;

  // ---- stage 1: gates (uses cur embeddings) ----
  for (int l = 0; l < L; ++l) {
    const float* W1a = edge_w1 + (size_t)l * 128 * 64;         // rows 0..63
    const float* W1b = W1a + (size_t)64 * 64;                  // rows 64..127
    gemm_cur64<false, false><<<gTile, 128, 0, stream>>>(cur_u, cur_i, U, Nn, W1a, nullptr, P);
    gemm_cur64<false, false><<<gTile, 128, 0, stream>>>(cur_u, cur_i, U, Nn, W1b, nullptr, Q);
    edge_gate_k<<<gEdge, 256, 0, stream>>>(P, Q, edge_b1 + (size_t)l * 64,
                                           edge_w2 + (size_t)l * 64, edge_b2 + l,
                                           row, col, adj, E, (unsigned)(2 * l),
                                           NE + (size_t)l * E);
    // HN = relu(cur @ node_w1[l] + node_b1[l])  (reuse P)
    gemm_cur64<true, true><<<gTile, 128, 0, stream>>>(cur_u, cur_i, U, Nn,
                                                      node_w1 + (size_t)l * 64 * 64,
                                                      node_b1 + (size_t)l * 64, P);
    node_gate_k<<<gNode, 256, 0, stream>>>(P, node_w2 + (size_t)l * 64, node_b2 + l,
                                           Nn, (unsigned)(2 * l + 1), NM + (size_t)l * Nn);
  }

  // ---- stage 2: edge-dropout view (uses orig embeddings) ----
  init_out_k<<<gN64, TB, 0, stream>>>(orig_u, orig_i, U, Nn, out, X);
  {
    float* xs = X; float* xd = XN;
    for (int l = 0; l < L; ++l) {
      zero_k<<<gN64, TB, 0, stream>>>(xd, ND);
      scatter_k<<<gEdge, 256, 0, stream>>>(row, col, NE + (size_t)l * E, xs, xd, E);
      accum_k<<<gN64, TB, 0, stream>>>(out, xd, Nn, 0);
      float* t = xs; xs = xd; xd = t;
    }
  }

  // ---- stage 3: node-dropout view ----
  copy_orig_k<<<gN64, TB, 0, stream>>>(orig_u, orig_i, U, Nn, X);
  {
    float* xs = X; float* xd = XN;
    for (int l = 0; l < L; ++l) {
      pool_mix_k<<<gNode, 256, 0, stream>>>(xs, rw_cols, rw_vals,
                                            NM + (size_t)l * Nn, Nn, K, MP);
      zero_k<<<gN64, TB, 0, stream>>>(xd, ND);
      scatter_k<<<gEdge, 256, 0, stream>>>(row, col, adj, MP, xd, E);
      accum_k<<<gN64, TB, 0, stream>>>(out, xd, Nn, 64);
      float* t = xs; xs = xd; xd = t;
    }
  }

  // ---- finalize: /(L+1) and reg scalars ----
  scale_k<<<gN128, TB, 0, stream>>>(out, ND * 2, 1.0f / (float)(L + 1));
  reduce_sum_k<<<256, 256, 0, stream>>>(NE, (size_t)L * E, 1.0f / ((float)E * (float)L),
                                        out + ND * 2);
  reduce_sum_k<<<256, 256, 0, stream>>>(NM, (size_t)L * Nn, 1.0f / ((float)Nn * (float)L),
                                        out + ND * 2 + 1);
}
